// cmil_mnist_70798240907693
// MI455X (gfx1250) — matrix operand order repaired and checked by reading, not compiled, not run on hardware
//
#include <hip/hip_runtime.h>

// ---------------- types ----------------
typedef __attribute__((ext_vector_type(16))) __bf16 v16bf;
typedef __attribute__((ext_vector_type(8)))  float  v8f;

__device__ __forceinline__ v8f wmma_bf16(v16bf a, v16bf b, v8f c) {
  // D = A(16x32) * B(32x16) + C ; f32 accumulate
  return __builtin_amdgcn_wmma_f32_16x16x32_bf16(false, a, false, b, (short)0, c, false, false);
}

// A-matrix (16x32, 16-bit) per-lane K offset for element e (contiguous: k = 16*(lane/16) + e)
__device__ __forceinline__ int ka_off(int e, int lane) {
  return e + (lane < 16 ? 0 : 16);
}

#define NINST 16384
#define NBAG  512
#define PER   32
#define IMG   2352

// ---------------- ws layout (bytes) ----------------
#define WB_CONV1W  ((size_t)0)           // bf16 [32*32]  (K padded 27->32)
#define WB_CONV2W  ((size_t)2048)        // bf16 [32*288]
#define WB_CONV3W  ((size_t)20480)       // bf16 [32*288]
#define WB_MUVW    ((size_t)38912)       // bf16 [128*512] (mu rows 0..63, lv rows 64..127)
#define WB_DFCW    ((size_t)169984)      // bf16 [512*64]
#define WB_H       ((size_t)235520)      // bf16 [16384*512]
#define WB_Z       ((size_t)17012736)    // f32  [16384*64]
#define WB_LOC     ((size_t)21207040)    // f32  [16384]
#define WB_YHAT    ((size_t)21272576)    // f32  [512]
#define WB_FLAT    ((size_t)21274624)    // i32  [512]
#define WB_MAXZB   ((size_t)21276672)    // bf16 [512*64]
#define WB_D0      ((size_t)21342208)    // f32  [512*512]
#define WB_D1      ((size_t)22390784)    // f32  [512*32*7*7]
#define WB_D2      ((size_t)25602048)    // f32  [512*32*14*14]
#define WB_KLP     ((size_t)38447104)    // f32  [1024]
#define WB_RECP    ((size_t)38451200)    // f32  [512]

// ---------------- kernel 0: pack weights to bf16 ----------------
__global__ __launch_bounds__(256) void pack_kernel(
    const float* __restrict__ w1, const float* __restrict__ w2,
    const float* __restrict__ w3, const float* __restrict__ muw,
    const float* __restrict__ lvw, const float* __restrict__ dfcw,
    __bf16* __restrict__ w1b, __bf16* __restrict__ w2b, __bf16* __restrict__ w3b,
    __bf16* __restrict__ muvb, __bf16* __restrict__ dfcb)
{
  int i = blockIdx.x * 256 + threadIdx.x;
  if (i < 1024) {                       // [32][32], zero-pad K 27..31
    int o = i >> 5, k = i & 31;
    w1b[i] = (k < 27) ? (__bf16)w1[o * 27 + k] : (__bf16)0.f;
  }
  if (i < 9216)  { w2b[i] = (__bf16)w2[i]; w3b[i] = (__bf16)w3[i]; }
  if (i < 32768) {
    muvb[i]         = (__bf16)muw[i];
    muvb[32768 + i] = (__bf16)lvw[i];
    dfcb[i]         = (__bf16)dfcw[i];
  }
}

// ---------------- kernel A: fused all-WMMA encoder, one instance per block ----------------
__global__ __launch_bounds__(256) void enc_kernel(
    const float* __restrict__ bag,
    const __bf16* __restrict__ w1b, const float* __restrict__ b1,
    const float* __restrict__ b2, const float* __restrict__ b3,
    const __bf16* __restrict__ w2b, const __bf16* __restrict__ w3b,
    __bf16* __restrict__ H)
{
  const int inst = blockIdx.x;
  const int tid  = threadIdx.x;
  const int lane = tid & 31;
  const int wave = tid >> 5;

  // overlay buffer: phase1 = xin(9408B f32) + bm1(13312B bf16 @ +9472)
  //                 phase2 = bm2 (36864B)   phase3 = bm3 (9216B)
  __shared__ __align__(32) char buf[36864];
  __shared__ __bf16 h1[32 * 196];   // 14x14
  __shared__ __bf16 h2[32 * 49];    // 7x7
  __shared__ __bf16 h3[512];        // 4x4 flattened co*16 + oy*4 + ox

  float*  xin = (float*)buf;
  __bf16* bm1 = (__bf16*)(buf + 9472);
  __bf16* bm  = (__bf16*)buf;

  const float* xg = bag + (size_t)inst * IMG;
  for (int i = tid; i < IMG; i += 256) xin[i] = xg[i];
  // warm conv2/conv3 weight rows in cache (global_prefetch_b8)
  if (wave == 0) {
    __builtin_prefetch(w2b + lane * 288, 0, 0);
    __builtin_prefetch(w3b + lane * 288, 0, 0);
  }
  __syncthreads();

  // --- build conv1 B tiles: im2col [K=32][N=208] in fragment order ---
  // idx = (ntile*32 + lane)*16 + e ; kk = (lane&16) + e ; p = ntile*16 + (lane&15)
  for (int f = tid; f < 208 * 32; f += 256) {
    int p = f >> 5, kk = f & 31;
    int idx = ((p >> 4) * 32 + (kk & 16) + (p & 15)) * 16 + (kk & 15);
    int ci = kk / 9, r = kk % 9, kh = r / 3, kw = r % 3;
    int oy = p / 14, ox = p % 14;
    int iy = 2 * oy + kh - 1, ix = 2 * ox + kw - 1;
    bool ok = (kk < 27) & (p < 196) & ((unsigned)iy < 28u) & ((unsigned)ix < 28u);
    int addr = ok ? (ci * 784 + iy * 28 + ix) : 0;
    float v = xin[addr];
    bm1[idx] = ok ? (__bf16)v : (__bf16)0.f;
  }
  __syncthreads();

  // --- conv1 (3->32, 28->14): 26 WMMA tiles over 8 waves ---
  {
    const int mtile = wave & 1;               // invariant: (wave + 8i) & 1 == wave & 1
    v16bf a;
#pragma unroll
    for (int e = 0; e < 16; ++e)
      a[e] = w1b[(mtile * 16 + (lane & 15)) * 32 + ka_off(e, lane)];
    for (int i = 0; i < 4; ++i) {
      int tile = wave + 8 * i;
      if (tile < 26) {
        int ntile = tile >> 1;
        v16bf b = *(const v16bf*)&bm1[(ntile * 32 + lane) * 16];
        v8f acc = {0.f, 0.f, 0.f, 0.f, 0.f, 0.f, 0.f, 0.f};
        acc = wmma_bf16(a, b, acc);
#pragma unroll
        for (int r = 0; r < 8; ++r) {
          int co = mtile * 16 + r + (lane < 16 ? 0 : 8);
          int pp = ntile * 16 + (lane & 15);
          if (pp < 196) {
            float v = acc[r] + b1[co];
            h1[co * 196 + pp] = (__bf16)(v > 0.f ? v : 0.f);
          }
        }
      }
    }
  }
  __syncthreads();

  // --- build conv2 B tiles: im2col [K=288][N=64] in fragment order ---
  // idx = ((ks*4 + ntile)*32 + lane)*16 + e
  for (int f = tid; f < 288 * 64; f += 256) {
    int kk = f >> 6, p = f & 63;
    int idx = (((kk >> 5) * 4 + (p >> 4)) * 32 + (kk & 16) + (p & 15)) * 16 + (kk & 15);
    int ci = kk / 9, r = kk % 9, kh = r / 3, kw = r % 3;
    int oy = p / 7, ox = p % 7;
    int iy = 2 * oy + kh - 1, ix = 2 * ox + kw - 1;
    bool ok = (p < 49) & ((unsigned)iy < 14u) & ((unsigned)ix < 14u);
    int addr = ok ? (ci * 196 + iy * 14 + ix) : 0;
    __bf16 v = h1[addr];
    bm[idx] = ok ? v : (__bf16)0.f;
  }
  __syncthreads();

  // --- conv2 (32->32, 14->7): M=32, N=49(pad 64), K=288 ---
  {
    const int mtile = wave & 1;
    const int ntile = wave >> 1;
    v8f acc = {0.f, 0.f, 0.f, 0.f, 0.f, 0.f, 0.f, 0.f};
    const int arow = mtile * 16 + (lane & 15);
    for (int ks = 0; ks < 9; ++ks) {
      v16bf a, b;
#pragma unroll
      for (int e = 0; e < 16; ++e)
        a[e] = w2b[arow * 288 + ks * 32 + ka_off(e, lane)];
      b = *(const v16bf*)&bm[((ks * 4 + ntile) * 32 + lane) * 16];
      acc = wmma_bf16(a, b, acc);
    }
#pragma unroll
    for (int r = 0; r < 8; ++r) {
      int co = mtile * 16 + r + (lane < 16 ? 0 : 8);
      int pp = ntile * 16 + (lane & 15);
      if (pp < 49) {
        float v = acc[r] + b2[co];
        h2[co * 49 + pp] = (__bf16)(v > 0.f ? v : 0.f);
      }
    }
  }
  __syncthreads();

  // --- build conv3 B tiles: im2col [K=288][N=16] in fragment order ---
  // idx = (ks*32 + lane)*16 + e
  for (int f = tid; f < 288 * 16; f += 256) {
    int kk = f >> 4, p = f & 15;
    int idx = ((kk >> 5) * 32 + (kk & 16) + p) * 16 + (kk & 15);
    int ci = kk / 9, r = kk % 9, kh = r / 3, kw = r % 3;
    int oy = p >> 2, ox = p & 3;
    int iy = 2 * oy + kh - 1, ix = 2 * ox + kw - 1;
    bool ok = ((unsigned)iy < 7u) & ((unsigned)ix < 7u);
    int addr = ok ? (ci * 49 + iy * 7 + ix) : 0;
    __bf16 v = h2[addr];
    bm[idx] = ok ? v : (__bf16)0.f;
  }
  __syncthreads();

  // --- conv3 (32->32, 7->4): M=32, N=16 (one tile), K=288 ---
  if (wave < 2) {
    const int mtile = wave;
    v8f acc = {0.f, 0.f, 0.f, 0.f, 0.f, 0.f, 0.f, 0.f};
    const int arow = mtile * 16 + (lane & 15);
    for (int ks = 0; ks < 9; ++ks) {
      v16bf a, b;
#pragma unroll
      for (int e = 0; e < 16; ++e)
        a[e] = w3b[arow * 288 + ks * 32 + ka_off(e, lane)];
      b = *(const v16bf*)&bm[(ks * 32 + lane) * 16];
      acc = wmma_bf16(a, b, acc);
    }
#pragma unroll
    for (int r = 0; r < 8; ++r) {
      int co = mtile * 16 + r + (lane < 16 ? 0 : 8);
      float v = acc[r] + b3[co];
      h3[co * 16 + (lane & 15)] = (__bf16)(v > 0.f ? v : 0.f);
    }
  }
  __syncthreads();

  for (int i = tid; i < 512; i += 256) H[(size_t)inst * 512 + i] = h3[i];
}

// ---------------- kernel B: mu/logvar GEMM + z + KL partial + loc ----------------
__global__ __launch_bounds__(256) void muvz_kernel(
    const __bf16* __restrict__ H, const __bf16* __restrict__ muvw,
    const float* __restrict__ mub, const float* __restrict__ lvb,
    const float* __restrict__ eps, const float* __restrict__ insw,
    const float* __restrict__ insb,
    float* __restrict__ z, float* __restrict__ loc, float* __restrict__ klp)
{
  __shared__ float mulv[16 * 128];
  __shared__ float zl[16 * 64];
  __shared__ float red[256];
  const int tid = threadIdx.x, lane = tid & 31, wave = tid >> 5;
  const int I0 = blockIdx.x * 16;

  v8f acc = {0.f, 0.f, 0.f, 0.f, 0.f, 0.f, 0.f, 0.f};
  const int arow = I0 + (lane & 15);
  const int outc = wave * 16 + (lane & 15);
  for (int ks = 0; ks < 16; ++ks) {
    const int K0 = ks * 32;
    v16bf a, b;
#pragma unroll
    for (int e = 0; e < 16; ++e)
      a[e] = H[(size_t)arow * 512 + K0 + ka_off(e, lane)];
#pragma unroll
    for (int e = 0; e < 16; ++e)
      b[e] = muvw[(size_t)outc * 512 + K0 + e + (lane < 16 ? 0 : 16)];
    acc = wmma_bf16(a, b, acc);
  }
#pragma unroll
  for (int r = 0; r < 8; ++r) {
    int mm = r + (lane < 16 ? 0 : 8);
    float bias = (outc < 64) ? mub[outc] : lvb[outc - 64];
    mulv[mm * 128 + outc] = acc[r] + bias;
  }
  __syncthreads();

  float kl = 0.f;
  for (int it = tid; it < 16 * 64; it += 256) {
    int is = it >> 6, j = it & 63;
    float mu = mulv[is * 128 + j];
    float lv = mulv[is * 128 + 64 + j];
    float sd = expf(0.5f * lv);
    float zz = mu + sd * eps[(size_t)(I0 + is) * 64 + j];
    zl[is * 64 + j] = zz;
    z[(size_t)(I0 + is) * 64 + j] = zz;
    kl += mu * mu + sd * sd - lv - 1.f;
  }
  red[tid] = kl;
  __syncthreads();
  for (int off = 128; off > 0; off >>= 1) {
    if (tid < off) red[tid] += red[tid + off];
    __syncthreads();
  }
  if (tid == 0) klp[blockIdx.x] = red[0];

  if (tid < 16) {
    float s = insb[0];
    for (int j = 0; j < 64; ++j) s += zl[tid * 64 + j] * insw[j];
    loc[I0 + tid] = s;
  }
}

// ---------------- kernel C: per-bag first-max argmax + gathers ----------------
__global__ __launch_bounds__(256) void bagmax_kernel(
    const float* __restrict__ loc, const float* __restrict__ z,
    float* __restrict__ yhat, int* __restrict__ flatidx, __bf16* __restrict__ maxzb)
{
  const int lane = threadIdx.x & 31, wave = threadIdx.x >> 5;
  const int g = blockIdx.x * 8 + wave;      // one wave32 per bag (PER==32)
  float v = loc[g * PER + lane];
  int idx = lane;
  for (int off = 16; off >= 1; off >>= 1) {
    float ov = __shfl_xor(v, off, 32);
    int   oi = __shfl_xor(idx, off, 32);
    if (ov > v || (ov == v && oi < idx)) { v = ov; idx = oi; }  // first-max tie-break
  }
  const int flat = g * PER + idx;
  if (lane == 0) { yhat[g] = v; flatidx[g] = flat; }
  for (int j = lane; j < 64; j += 32)
    maxzb[g * 64 + j] = (__bf16)z[(size_t)flat * 64 + j];
}

// ---------------- kernel D: decoder fc [512,64]@[64,512] via WMMA ----------------
__global__ __launch_bounds__(256) void decfc_kernel(
    const __bf16* __restrict__ maxzb, const __bf16* __restrict__ dfcwb,
    const float* __restrict__ dfcb, float* __restrict__ d0)
{
  const int lane = threadIdx.x & 31, wave = threadIdx.x >> 5;
  const int t = blockIdx.x * 8 + wave;      // 0..1023 tiles
  const int mt = t >> 5, nt = t & 31;
  v8f acc = {0.f, 0.f, 0.f, 0.f, 0.f, 0.f, 0.f, 0.f};
  for (int ks = 0; ks < 2; ++ks) {
    const int K0 = ks * 32;
    v16bf a, b;
#pragma unroll
    for (int e = 0; e < 16; ++e)
      a[e] = maxzb[(mt * 16 + (lane & 15)) * 64 + K0 + ka_off(e, lane)];
#pragma unroll
    for (int e = 0; e < 16; ++e)
      b[e] = dfcwb[(nt * 16 + (lane & 15)) * 64 + K0 + e + (lane < 16 ? 0 : 16)];
    acc = wmma_bf16(a, b, acc);
  }
#pragma unroll
  for (int r = 0; r < 8; ++r) {
    int bagi = mt * 16 + r + (lane < 16 ? 0 : 8);
    int unit = nt * 16 + (lane & 15);
    float v = acc[r] + dfcb[unit];
    d0[(size_t)bagi * 512 + unit] = v > 0.f ? v : 0.f;
  }
}

// ---------------- transposed conv (stride 2, pad 1) + relu; one bag per block ----------------
__global__ __launch_bounds__(256) void convt_kernel(
    const float* __restrict__ xin, const float* __restrict__ w,
    const float* __restrict__ bias, float* __restrict__ out,
    int Cin, int Cout, int IH, int OH)
{
  const int bagi = blockIdx.x;
  const float* xi = xin + (size_t)bagi * Cin * IH * IH;
  float* yo = out + (size_t)bagi * Cout * OH * OH;
  const int tot = Cout * OH * OH;
  for (int o = threadIdx.x; o < tot; o += 256) {
    int co = o / (OH * OH), rem = o % (OH * OH), oy = rem / OH, ox = rem % OH;
    float acc = bias[co];
    for (int ky = 0; ky < 3; ++ky) {
      int t2 = oy + 1 - ky;
      if (t2 < 0 || (t2 & 1)) continue;
      int iy = t2 >> 1;
      if (iy >= IH) continue;
      for (int kx = 0; kx < 3; ++kx) {
        int s2 = ox + 1 - kx;
        if (s2 < 0 || (s2 & 1)) continue;
        int ix = s2 >> 1;
        if (ix >= IH) continue;
        const int wb = co * 9 + ky * 3 + kx;
        for (int ci = 0; ci < Cin; ++ci)
          acc += xi[ci * IH * IH + iy * IH + ix] * w[ci * Cout * 9 + wb];
      }
    }
    yo[o] = acc > 0.f ? acc : 0.f;
  }
}

// ---------------- final convT + sigmoid + MSE partial per bag ----------------
__global__ __launch_bounds__(256) void convt3_mse_kernel(
    const float* __restrict__ d2, const float* __restrict__ w,
    const float* __restrict__ bias, const float* __restrict__ bag,
    const int* __restrict__ flatidx, float* __restrict__ recp)
{
  __shared__ float red[256];
  const int bagi = blockIdx.x;
  const float* xi = d2 + (size_t)bagi * 32 * 196;    // 32x14x14
  const float* xt = bag + (size_t)flatidx[bagi] * IMG;
  float local = 0.f;
  for (int o = threadIdx.x; o < IMG; o += 256) {     // 3x28x28
    int co = o / 784, rem = o % 784, oy = rem / 28, ox = rem % 28;
    float acc = bias[co];
    for (int ky = 0; ky < 3; ++ky) {
      int t2 = oy + 1 - ky;
      if (t2 < 0 || (t2 & 1)) continue;
      int iy = t2 >> 1;
      if (iy >= 14) continue;
      for (int kx = 0; kx < 3; ++kx) {
        int s2 = ox + 1 - kx;
        if (s2 < 0 || (s2 & 1)) continue;
        int ix = s2 >> 1;
        if (ix >= 14) continue;
        const int wb = co * 9 + ky * 3 + kx;
        for (int ci = 0; ci < 32; ++ci)
          acc += xi[ci * 196 + iy * 14 + ix] * w[ci * 27 + wb];
      }
    }
    float img = 1.f / (1.f + expf(-acc));
    float d = img - xt[o];
    local += d * d;
  }
  red[threadIdx.x] = local;
  __syncthreads();
  for (int off = 128; off > 0; off >>= 1) {
    if (threadIdx.x < off) red[threadIdx.x] += red[threadIdx.x + off];
    __syncthreads();
  }
  if (threadIdx.x == 0) recp[bagi] = red[0];
}

// ---------------- final deterministic reductions -> [recon, KL, aux] ----------------
__global__ __launch_bounds__(256) void final_kernel(
    const float* __restrict__ recp, const float* __restrict__ klp,
    const float* __restrict__ yhat, const float* __restrict__ label,
    float* __restrict__ out)
{
  __shared__ float red[256];
  const int t = threadIdx.x;

  red[t] = recp[t] + recp[t + 256];
  __syncthreads();
  for (int off = 128; off > 0; off >>= 1) {
    if (t < off) red[t] += red[t + off];
    __syncthreads();
  }
  if (t == 0) out[0] = red[0] / (512.f * 2352.f);
  __syncthreads();

  red[t] = klp[t] + klp[t + 256] + klp[t + 512] + klp[t + 768];
  __syncthreads();
  for (int off = 128; off > 0; off >>= 1) {
    if (t < off) red[t] += red[t + off];
    __syncthreads();
  }
  if (t == 0) out[1] = 0.5f * red[0] / (16384.f * 64.f);
  __syncthreads();

  float s = 0.f;
#pragma unroll
  for (int k = 0; k < 2; ++k) {
    int i = t + k * 256;
    float y = yhat[i], tg = label[i];
    s += fmaxf(y, 0.f) - y * tg + log1pf(expf(-fabsf(y)));
  }
  red[t] = s;
  __syncthreads();
  for (int off = 128; off > 0; off >>= 1) {
    if (t < off) red[t] += red[t + off];
    __syncthreads();
  }
  if (t == 0) out[2] = red[0] / 512.f;
}

// ---------------- host launcher ----------------
extern "C" void kernel_launch(void* const* d_in, const int* in_sizes, int n_in,
                              void* d_out, int out_size, void* d_ws, size_t ws_size,
                              hipStream_t stream) {
  const float* bag       = (const float*)d_in[0];
  // d_in[1]: bag_idx (unused: bags are contiguous arange(N)//PER)
  const float* bag_label = (const float*)d_in[2];
  const float* eps       = (const float*)d_in[3];
  const float* enc_w1 = (const float*)d_in[4];
  const float* enc_b1 = (const float*)d_in[5];
  const float* enc_w2 = (const float*)d_in[6];
  const float* enc_b2 = (const float*)d_in[7];
  const float* enc_w3 = (const float*)d_in[8];
  const float* enc_b3 = (const float*)d_in[9];
  const float* mu_w   = (const float*)d_in[10];
  const float* mu_b   = (const float*)d_in[11];
  const float* lv_w   = (const float*)d_in[12];
  const float* lv_b   = (const float*)d_in[13];
  const float* ins_w  = (const float*)d_in[14];
  const float* ins_b  = (const float*)d_in[15];
  const float* dfc_w  = (const float*)d_in[16];
  const float* dfc_b  = (const float*)d_in[17];
  const float* de1_w  = (const float*)d_in[18];
  const float* de1_b  = (const float*)d_in[19];
  const float* de2_w  = (const float*)d_in[20];
  const float* de2_b  = (const float*)d_in[21];
  const float* de3_w  = (const float*)d_in[22];
  const float* de3_b  = (const float*)d_in[23];

  char* ws = (char*)d_ws;
  __bf16* w1b   = (__bf16*)(ws + WB_CONV1W);
  __bf16* w2b   = (__bf16*)(ws + WB_CONV2W);
  __bf16* w3b   = (__bf16*)(ws + WB_CONV3W);
  __bf16* muvb  = (__bf16*)(ws + WB_MUVW);
  __bf16* dfcwb = (__bf16*)(ws + WB_DFCW);
  __bf16* H     = (__bf16*)(ws + WB_H);
  float*  z     = (float*)(ws + WB_Z);
  float*  loc   = (float*)(ws + WB_LOC);
  float*  yhat  = (float*)(ws + WB_YHAT);
  int*    flat  = (int*)(ws + WB_FLAT);
  __bf16* maxzb = (__bf16*)(ws + WB_MAXZB);
  float*  d0    = (float*)(ws + WB_D0);
  float*  d1    = (float*)(ws + WB_D1);
  float*  d2    = (float*)(ws + WB_D2);
  float*  klp   = (float*)(ws + WB_KLP);
  float*  recp  = (float*)(ws + WB_RECP);

  pack_kernel<<<128, 256, 0, stream>>>(enc_w1, enc_w2, enc_w3, mu_w, lv_w, dfc_w,
                                       w1b, w2b, w3b, muvb, dfcwb);
  enc_kernel<<<NINST, 256, 0, stream>>>(bag, w1b, enc_b1, enc_b2, enc_b3,
                                        w2b, w3b, H);
  muvz_kernel<<<NINST / 16, 256, 0, stream>>>(H, muvb, mu_b, lv_b, eps,
                                              ins_w, ins_b, z, loc, klp);
  bagmax_kernel<<<NBAG / 8, 256, 0, stream>>>(loc, z, yhat, flat, maxzb);
  decfc_kernel<<<128, 256, 0, stream>>>(maxzb, dfcwb, dfc_b, d0);
  convt_kernel<<<NBAG, 256, 0, stream>>>(d0, de1_w, de1_b, d1, 32, 32, 4, 7);
  convt_kernel<<<NBAG, 256, 0, stream>>>(d1, de2_w, de2_b, d2, 32, 32, 7, 14);
  convt3_mse_kernel<<<NBAG, 256, 0, stream>>>(d2, de3_w, de3_b, bag, flat, recp);
  final_kernel<<<1, 256, 0, stream>>>(recp, klp, yhat, bag_label, (float*)d_out);
}
